// MultiHeadAttention_81114752352880
// MI455X (gfx1250) — compile-verified, matrix operand order checked by reading, not run on hardware
//
#include <hip/hip_runtime.h>
#include <hip/hip_bf16.h>

// ---------------------------------------------------------------------------
// MI455X (gfx1250) multi-head attention, bf16 WMMA path, wave32.
// B=4, T=2048, C=1024, H=16, D=64.
// - All WMMA operands are A-style row-major fragments = 2 x global_load_b128
//   (W pre-transposed once; V produced transposed by its projection).
// - Softmax in base-2 with fused stats C = m + log2(l):  p = 2^(s*c2 - C),
//   using v_exp_f32 / v_log_f32 directly (hardware is base-2 native).
// - Rowstats: two sweeps (register max, then exp2 sums) -> cross-lane
//   reduction only once per wave instead of per k-tile.
// - K chunks in P@V staged to LDS with global_load_async_to_lds_b128,
//   double-buffered on s_wait_asynccnt.
// ---------------------------------------------------------------------------

typedef __bf16 bf16_t;
typedef __attribute__((ext_vector_type(16))) __bf16        v16bf;
typedef __attribute__((ext_vector_type(4)))  __bf16        v4bf;
typedef __attribute__((ext_vector_type(8)))  float         v8f;
typedef __attribute__((ext_vector_type(4)))  unsigned int  u32x4;
typedef __attribute__((ext_vector_type(4)))  int           i32x4;

#define BB 4
#define TT 2048
#define CC 1024
#define HH 16
#define DD 64

// scale2 = (1/sqrt(D)) * log2(e)
#define SCALE2 0.1803368801111f

__device__ __forceinline__ float fast_exp2(float x) {
  return __builtin_amdgcn_exp2f(x);   // v_exp_f32 (base-2 native)
}
__device__ __forceinline__ float fast_log2(float x) {
  return __builtin_amdgcn_logf(x);    // v_log_f32 (base-2 native)
}

// ---- optional CDNA5 async-to-LDS path (guarded; fallback = direct loads) ---
#if defined(__has_builtin)
#if __has_builtin(__builtin_amdgcn_global_load_async_to_lds_b128)
#define HAVE_ASYNC_LDS 1
#endif
#endif
#ifndef HAVE_ASYNC_LDS
#define HAVE_ASYNC_LDS 0
#endif

#if HAVE_ASYNC_LDS
// builtin signature (from hipcc diagnostic): (v4i as1*, v4i as3*, imm, imm)
#define GPTR(p) ((__attribute__((address_space(1))) i32x4*)(p))
#define LPTR(p) ((__attribute__((address_space(3))) i32x4*)(p))
#if __has_builtin(__builtin_amdgcn_s_wait_asynccnt)
#define WAIT_ASYNC(n) __builtin_amdgcn_s_wait_asynccnt(n)
#else
#define WAIT_ASYNC(n) asm volatile("s_wait_asynccnt " #n ::: "memory")
#endif
#endif

__device__ __forceinline__ v8f wmma_bf16(v16bf a, v16bf b, v8f c) {
  // D = A(16x32) * B(32x16) + C, f32 accumulate
  return __builtin_amdgcn_wmma_f32_16x16x32_bf16(
      /*neg_a=*/false, a, /*neg_b=*/false, b,
      /*c_mod=*/(short)0, c, /*reuse_a=*/false, /*reuse_b=*/false);
}

// A-style fragment (16x32 MxK or, transpose-symmetric, 32x16 KxN from a
// pre-transposed source). ISA layout: lane L holds row m=L%16, VGPR v holds
// k-pair k = (L<16?0:8) + 2*(v&3) + 16*(v>>2)  ==  two contiguous 16B runs:
// [kb, kb+8) and [kb+16, kb+24).  => exactly 2 x b128 loads per lane.
struct frag_bits { u32x4 lo, hi; };

__device__ __forceinline__ v16bf load_frag(const bf16_t* A, int lda, int lane) {
  int m  = lane & 15;
  int kb = (lane & 16) ? 8 : 0;
  const bf16_t* row = A + (size_t)m * lda + kb;
  frag_bits t;
  t.lo = *(const u32x4*)(row);
  t.hi = *(const u32x4*)(row + 16);
  return __builtin_bit_cast(v16bf, t);
}

// ---------------------------------------------------------------------------
// Kernel 1: f32 -> bf16 cast, 4 elements/thread (v_cvt_pk_bf16_f32 pairs)
// ---------------------------------------------------------------------------
__global__ void cast_bf16_kernel(const float4* __restrict__ src,
                                 v4bf* __restrict__ dst, int n4) {
  int i = blockIdx.x * blockDim.x + threadIdx.x;
  if (i < n4) {
    float4 f = src[i];
    v4bf o;
    o[0] = (bf16_t)f.x; o[1] = (bf16_t)f.y;
    o[2] = (bf16_t)f.z; o[3] = (bf16_t)f.w;
    dst[i] = o;
  }
}

// ---------------------------------------------------------------------------
// Kernel 2: fused cast + transpose for weights: dst[n*C + k] = (bf16)src[k*C+n]
// ---------------------------------------------------------------------------
__global__ __launch_bounds__(256) void cast_transpose_bf16(
    const float* __restrict__ src, bf16_t* __restrict__ dst) {
  __shared__ float tile[32][33];
  int tx = threadIdx.x & 31;
  int ty = threadIdx.x >> 5;  // 0..7
#pragma unroll
  for (int i = 0; i < 4; ++i) {
    int k = blockIdx.y * 32 + ty * 4 + i;
    int n = blockIdx.x * 32 + tx;
    tile[ty * 4 + i][tx] = src[(size_t)k * CC + n];
  }
  __syncthreads();
#pragma unroll
  for (int i = 0; i < 4; ++i) {
    int n = blockIdx.x * 32 + ty * 4 + i;
    int k = blockIdx.y * 32 + tx;
    dst[(size_t)n * CC + k] = (bf16_t)tile[tx][ty * 4 + i];
  }
}

// ---------------------------------------------------------------------------
// Kernel 3: GEMM + bias. One wave: 16x64 tile; per k-step 2 b128 A loads
// (reused 4x) + 4x2 b128 B loads + 4 WMMA.
// MODE 0: bf16 out, head-split (B,H,T,D)      (Q, K projections)
// MODE 1: bf16 out, head-split (B,H,D,T)      (V projection, pre-transposed)
// MODE 2: f32  out, row-major (B*T, C)        (output projection)
// ---------------------------------------------------------------------------
template <int MODE>
__global__ __launch_bounds__(256) void gemm16x64(
    const bf16_t* __restrict__ X, const bf16_t* __restrict__ WT,
    const float* __restrict__ bias, void* __restrict__ outp) {
  const int K = CC;
  int lane = threadIdx.x & 31;
  int wave = blockIdx.x * (blockDim.x >> 5) + (threadIdx.x >> 5);
  int tm = wave >> 4;  // 512 row tiles
  int tn = wave & 15;  // 16 col super-tiles (64 wide)
  const bf16_t* Ap = X + (size_t)tm * 16 * K;

  v8f zero = {};
  v8f acc[4] = {zero, zero, zero, zero};
  for (int k0 = 0; k0 < K; k0 += 32) {
    if (k0 + 64 < K) __builtin_prefetch(Ap + k0 + 64, 0, 3);
    v16bf a = load_frag(Ap + k0, K, lane);
#pragma unroll
    for (int nt = 0; nt < 4; ++nt) {
      const bf16_t* Bp = WT + (size_t)(tn * 64 + nt * 16) * K + k0;
      v16bf b = load_frag(Bp, K, lane);
      acc[nt] = wmma_bf16(a, b, acc[nt]);
    }
  }
  int n_ = lane & 15;
  int mo = (lane & 16) ? 8 : 0;
#pragma unroll
  for (int nt = 0; nt < 4; ++nt) {
    int ncol = tn * 64 + nt * 16 + n_;
    float bv = bias[ncol];
    int h = ncol >> 6, d = ncol & 63;
#pragma unroll
    for (int r = 0; r < 8; ++r) {
      int mrow = tm * 16 + r + mo;
      float val = acc[nt][r] + bv;
      if (MODE == 0) {
        int b_ = mrow >> 11, t = mrow & (TT - 1);
        ((bf16_t*)outp)[(((size_t)b_ * HH + h) * TT + t) * DD + d] =
            (bf16_t)val;
      } else if (MODE == 1) {
        int b_ = mrow >> 11, t = mrow & (TT - 1);
        ((bf16_t*)outp)[(((size_t)b_ * HH + h) * DD + d) * TT + t] =
            (bf16_t)val;
      } else {
        ((float*)outp)[(size_t)mrow * CC + ncol] = val;
      }
    }
  }
}

// ---------------------------------------------------------------------------
// Kernel 4: softmax stats, two sweeps. Writes C = m2 + log2(sum) per row,
// where m2 = max(s * SCALE2).  probs = 2^(s*SCALE2 - C).
// Sweep 1: register max only (no shuffles). One cross-lane reduction.
// Sweep 2: register exp2 sums. One cross-lane reduction.
// ---------------------------------------------------------------------------
__global__ __launch_bounds__(256) void attn_rowstats(
    const bf16_t* __restrict__ Q, const bf16_t* __restrict__ Kh,
    float* __restrict__ Crow) {
  int lane = threadIdx.x & 31;
  int wave = blockIdx.x * (blockDim.x >> 5) + (threadIdx.x >> 5);
  int bh = wave >> 7;
  int qt = wave & 127;
  const bf16_t* Qp = Q + (size_t)bh * TT * DD + (size_t)qt * 16 * DD;
  const bf16_t* Kp = Kh + (size_t)bh * TT * DD;
  v16bf q0 = load_frag(Qp, DD, lane);
  v16bf q1 = load_frag(Qp + 32, DD, lane);

  // ---- sweep 1: per-lane raw max ----
  float pm[8];
#pragma unroll
  for (int r = 0; r < 8; ++r) pm[r] = -3.0e38f;
  for (int j = 0; j < 128; ++j) {
    const bf16_t* Kj = Kp + (size_t)j * 16 * DD;
    v16bf k0 = load_frag(Kj, DD, lane);       // B-frag of K^T == A-frag of K
    v16bf k1 = load_frag(Kj + 32, DD, lane);
    v8f s = {};
    s = wmma_bf16(q0, k0, s);
    s = wmma_bf16(q1, k1, s);
#pragma unroll
    for (int r = 0; r < 8; ++r) pm[r] = fmaxf(pm[r], s[r]);
  }
  // one cross-lane max per register (masks 1,2,4,8 stay inside 16-lane half)
#pragma unroll
  for (int r = 0; r < 8; ++r) {
#pragma unroll
    for (int off = 1; off < 16; off <<= 1)
      pm[r] = fmaxf(pm[r], __shfl_xor(pm[r], off, 32));
    pm[r] *= SCALE2;  // m2 = max of scaled scores (SCALE2 > 0)
  }

  // ---- sweep 2: per-lane exp2 sums against known max ----
  float ps[8];
#pragma unroll
  for (int r = 0; r < 8; ++r) ps[r] = 0.0f;
  for (int j = 0; j < 128; ++j) {
    const bf16_t* Kj = Kp + (size_t)j * 16 * DD;
    v16bf k0 = load_frag(Kj, DD, lane);
    v16bf k1 = load_frag(Kj + 32, DD, lane);
    v8f s = {};
    s = wmma_bf16(q0, k0, s);
    s = wmma_bf16(q1, k1, s);
#pragma unroll
    for (int r = 0; r < 8; ++r)
      ps[r] += fast_exp2(fmaf(s[r], SCALE2, -pm[r]));
  }
#pragma unroll
  for (int r = 0; r < 8; ++r) {
#pragma unroll
    for (int off = 1; off < 16; off <<= 1)
      ps[r] += __shfl_xor(ps[r], off, 32);
  }
  if ((lane & 15) == 0) {
    int mo = (lane & 16) ? 8 : 0;
#pragma unroll
    for (int r = 0; r < 8; ++r) {
      int q = qt * 16 + r + mo;
      Crow[(size_t)bh * TT + q] = pm[r] + fast_log2(ps[r]);
    }
  }
}

// ---------------------------------------------------------------------------
// Kernel 5: avg_probs. Wave owns exclusive (b, qtile, 128-col slice) region,
// loops all 16 heads (deterministic, no atomics). 1/H folded into C (+4).
// ---------------------------------------------------------------------------
__global__ __launch_bounds__(256) void attn_avg(
    const bf16_t* __restrict__ Q, const bf16_t* __restrict__ Kh,
    const float* __restrict__ Crow, float* __restrict__ avg) {
  int lane = threadIdx.x & 31;
  int wave = blockIdx.x * (blockDim.x >> 5) + (threadIdx.x >> 5);
  int b  = wave >> 11;
  int qt = (wave >> 4) & 127;
  int ks = wave & 15;
  int n_ = lane & 15;
  int mo = (lane & 16) ? 8 : 0;

  float acc[8][8];
#pragma unroll
  for (int j = 0; j < 8; ++j)
#pragma unroll
    for (int r = 0; r < 8; ++r) acc[j][r] = 0.0f;

  for (int h = 0; h < HH; ++h) {
    int bh = b * HH + h;
    const bf16_t* Qp = Q + (size_t)bh * TT * DD + (size_t)qt * 16 * DD;
    v16bf q0 = load_frag(Qp, DD, lane);
    v16bf q1 = load_frag(Qp + 32, DD, lane);
    float c4[8];
#pragma unroll
    for (int r = 0; r < 8; ++r) {
      int q = qt * 16 + r + mo;
      c4[r] = Crow[(size_t)bh * TT + q] + 4.0f;  // +log2(H) folds the 1/16
    }
    for (int j = 0; j < 8; ++j) {
      const bf16_t* Kj =
          Kh + (size_t)bh * TT * DD + (size_t)(ks * 128 + j * 16) * DD;
      v16bf k0 = load_frag(Kj, DD, lane);
      v16bf k1 = load_frag(Kj + 32, DD, lane);
      v8f s = {};
      s = wmma_bf16(q0, k0, s);
      s = wmma_bf16(q1, k1, s);
#pragma unroll
      for (int r = 0; r < 8; ++r)
        acc[j][r] += fast_exp2(fmaf(s[r], SCALE2, -c4[r]));
    }
  }
#pragma unroll
  for (int j = 0; j < 8; ++j)
#pragma unroll
    for (int r = 0; r < 8; ++r) {
      int q = qt * 16 + r + mo;
      int kc = ks * 128 + j * 16 + n_;
      avg[((size_t)b * TT + q) * TT + kc] = acc[j][r];
    }
}

// ---------------------------------------------------------------------------
// Kernel 6: out = P @ V per (b,h,qtile). K chunks (32x64 bf16 = 4 KB)
// double-buffer staged to LDS via global_load_async_to_lds_b128 +
// s_wait_asynccnt; probs bounce through LDS (C-layout -> A-layout, b16 writes, b128
// reads); V read directly from pre-transposed (B,H,D,T) layout.
// ---------------------------------------------------------------------------
__global__ __launch_bounds__(256) void attn_pv(
    const bf16_t* __restrict__ Q, const bf16_t* __restrict__ Kh,
    const bf16_t* __restrict__ Vt, const float* __restrict__ Crow,
    bf16_t* __restrict__ AO) {
  __shared__ __bf16 pbuf[8][16 * 32];         // per-wave probs tile (1 KB)
#if HAVE_ASYNC_LDS
  __shared__ __bf16 kbuf[8][2][32 * 64];      // per-wave double-buffered K
#endif
  int lane = threadIdx.x & 31;
  int w = threadIdx.x >> 5;
  int wave = blockIdx.x * (blockDim.x >> 5) + w;
  int bh = wave >> 7, qt = wave & 127;
  int n_ = lane & 15, mo = (lane & 16) ? 8 : 0;

  const bf16_t* Qp  = Q + (size_t)bh * TT * DD + (size_t)qt * 16 * DD;
  const bf16_t* Kp  = Kh + (size_t)bh * TT * DD;
  const bf16_t* Vtp = Vt + (size_t)bh * DD * TT;
  v16bf q0 = load_frag(Qp, DD, lane);
  v16bf q1 = load_frag(Qp + 32, DD, lane);
  float cr[8];
#pragma unroll
  for (int r = 0; r < 8; ++r) {
    int q = qt * 16 + r + mo;
    cr[r] = Crow[(size_t)bh * TT + q];
  }

  v8f zero = {};
  v8f o[4] = {zero, zero, zero, zero};

#if HAVE_ASYNC_LDS
  {  // prologue: stage chunk 0 into buffer 0 (8 async b128 ops, 512B each)
    char* s = (char*)Kp;
#pragma unroll
    for (int i = 0; i < 8; ++i) {
      size_t off = ((size_t)i * 32 + lane) * 16;
      __builtin_amdgcn_global_load_async_to_lds_b128(
          GPTR(s + off), LPTR((char*)&kbuf[w][0][0] + off), 0, 0);
    }
  }
#endif

  for (int j2 = 0; j2 < 64; ++j2) {  // 32 key positions per chunk
#if HAVE_ASYNC_LDS
    int cur = j2 & 1;
    if (j2 + 1 < 64) {
      char* s = (char*)(Kp + (size_t)(j2 + 1) * 32 * DD);
#pragma unroll
      for (int i = 0; i < 8; ++i) {
        size_t off = ((size_t)i * 32 + lane) * 16;
        __builtin_amdgcn_global_load_async_to_lds_b128(
            GPTR(s + off), LPTR((char*)&kbuf[w][cur ^ 1][0] + off), 0, 0);
      }
      WAIT_ASYNC(8);   // chunk j2 complete, j2+1 may be in flight
    } else {
      WAIT_ASYNC(0);
    }
    const bf16_t* Kc = (const bf16_t*)&kbuf[w][cur][0];
#else
    const bf16_t* Kc = Kp + (size_t)j2 * 32 * DD;
#endif
#pragma unroll
    for (int half = 0; half < 2; ++half) {
      const bf16_t* Kj = Kc + half * 16 * DD;
      v16bf k0 = load_frag(Kj, DD, lane);
      v16bf k1 = load_frag(Kj + 32, DD, lane);
      v8f s = {};
      s = wmma_bf16(q0, k0, s);
      s = wmma_bf16(q1, k1, s);
#pragma unroll
      for (int r = 0; r < 8; ++r) {
        float p = fast_exp2(fmaf(s[r], SCALE2, -cr[r]));
        pbuf[w][(r + mo) * 32 + half * 16 + n_] = (__bf16)p;
      }
    }
    // LDS is in-order within a wave; block compiler reordering only.
    asm volatile("" ::: "memory");
    v16bf pa = load_frag((const bf16_t*)&pbuf[w][0], 32, lane);  // 2x ds b128
#pragma unroll
    for (int nt = 0; nt < 4; ++nt) {
      const bf16_t* Vc = Vtp + (size_t)(nt * 16) * TT + (size_t)j2 * 32;
      v16bf vb = load_frag(Vc, TT, lane);  // 2x b128 from transposed V
      o[nt] = wmma_bf16(pa, vb, o[nt]);
    }
    asm volatile("" ::: "memory");
  }

  int b_ = bh >> 4, h = bh & 15;
#pragma unroll
  for (int nt = 0; nt < 4; ++nt)
#pragma unroll
    for (int r = 0; r < 8; ++r) {
      int q = qt * 16 + r + mo;
      AO[((size_t)b_ * TT + q) * CC + h * DD + nt * 16 + n_] = (bf16_t)o[nt][r];
    }
}

// ---------------------------------------------------------------------------
// Launcher
// ---------------------------------------------------------------------------
extern "C" void kernel_launch(void* const* d_in, const int* in_sizes, int n_in,
                              void* d_out, int out_size, void* d_ws,
                              size_t ws_size, hipStream_t stream) {
  const float* x  = (const float*)d_in[0];
  const float* Wq = (const float*)d_in[1];
  const float* bq = (const float*)d_in[2];
  const float* Wk = (const float*)d_in[3];
  const float* bk = (const float*)d_in[4];
  const float* Wv = (const float*)d_in[5];
  const float* bv = (const float*)d_in[6];
  const float* Wo = (const float*)d_in[7];
  const float* bo = (const float*)d_in[8];

  float* out = (float*)d_out;               // (B,T,C)
  float* avg = out + (size_t)BB * TT * CC;  // (B,T,T)

  const size_t MTC = (size_t)BB * TT * CC;  // 8,388,608
  const size_t WSZ = (size_t)CC * CC;       // 1,048,576

  // workspace carve (~93 MB)
  bf16_t* xb  = (bf16_t*)d_ws;
  bf16_t* wqT = xb + MTC;     // (N x K) transposed weights
  bf16_t* wkT = wqT + WSZ;
  bf16_t* wvT = wkT + WSZ;
  bf16_t* woT = wvT + WSZ;
  bf16_t* Qb  = woT + WSZ;    // (B,H,T,D)
  bf16_t* Kb  = Qb + MTC;     // (B,H,T,D)
  bf16_t* VbT = Kb + MTC;     // (B,H,D,T)  pre-transposed
  bf16_t* AOb = VbT + MTC;    // (B,T,C)
  float*  Cr  = (float*)(AOb + MTC);        // B*H*T fused softmax stats

  // 1) casts (x vectorized; weights fused cast+transpose)
  cast_bf16_kernel<<<(int)(MTC / 4 + 255) / 256, 256, 0, stream>>>(
      (const float4*)x, (v4bf*)xb, (int)(MTC / 4));
  dim3 tgrid(CC / 32, CC / 32);
  cast_transpose_bf16<<<tgrid, 256, 0, stream>>>(Wq, wqT);
  cast_transpose_bf16<<<tgrid, 256, 0, stream>>>(Wk, wkT);
  cast_transpose_bf16<<<tgrid, 256, 0, stream>>>(Wv, wvT);
  cast_transpose_bf16<<<tgrid, 256, 0, stream>>>(Wo, woT);

  // 2) projections
  gemm16x64<0><<<1024, 256, 0, stream>>>(xb, wqT, bq, Qb);
  gemm16x64<0><<<1024, 256, 0, stream>>>(xb, wkT, bk, Kb);
  gemm16x64<1><<<1024, 256, 0, stream>>>(xb, wvT, bv, VbT);

  // 3) softmax stats (two-sweep, fused C = m + log2(l))
  attn_rowstats<<<1024, 256, 0, stream>>>(Qb, Kb, Cr);

  // 4) head-averaged probs
  attn_avg<<<1024, 256, 0, stream>>>(Qb, Kb, Cr, avg);

  // 5) P @ V -> (B,T,C) bf16
  attn_pv<<<1024, 256, 0, stream>>>(Qb, Kb, VbT, Cr, AOb);

  // 6) output projection -> f32 d_out
  gemm16x64<2><<<1024, 256, 0, stream>>>(AOb, woT, bo, out);
}
